// MultiHeadSelfAttention_31301721653741
// MI455X (gfx1250) — matrix operand order repaired and checked by reading, not compiled, not run on hardware
//
#include <hip/hip_runtime.h>

// ---------------------------------------------------------------------------
// MHSA forward for MI455X (gfx1250, wave32, WMMA 16x16x32 bf16)
//   x:(4,2048,1024) f32, w_qkv:(3072,1024) f32, w_o:(1024,1024) f32 -> f32
// Pipeline:
//   0) one-pass f32->bf16 conversion of x / w_qkv / w_o   (kills inner-loop cvts)
//   1) QKV GEMM (bf16 x bf16 -> f32 acc), scatter to Q/K rows + V^T
//   2) flash attention, transposed, K/V staged in LDS via
//      global_load_async_to_lds_b128 (double buffered, s_wait_asynccnt)
//   3) output GEMM -> f32
// ---------------------------------------------------------------------------

typedef __attribute__((ext_vector_type(16))) __bf16 v16bf;
typedef __attribute__((ext_vector_type(8)))  __bf16 v8bf;
typedef __attribute__((ext_vector_type(4)))  __bf16 v4bf;
typedef __attribute__((ext_vector_type(8)))  float  v8f;
typedef __attribute__((ext_vector_type(4)))  float  v4f;

#define WMMA_BF16(a, b, c) \
  __builtin_amdgcn_wmma_f32_16x16x32_bf16(false, (a), false, (b), (short)0, (c), false, false)

// A operand (16x32 bf16): lane<16 holds row M=lane, K 0-15 contiguous; lane>=16
// holds same row, K 16-31.  Caller passes p = row_base + (lane>>4)*16.
__device__ __forceinline__ v16bf frag_a_bf16(const __bf16* __restrict__ p) {
  v8bf lo = *(const v8bf*)p;
  v8bf hi = *(const v8bf*)(p + 8);
  v16bf f;
#pragma unroll
  for (int i = 0; i < 8; ++i) { f[i] = lo[i]; f[i + 8] = hi[i]; }
  return f;
}
// B operand (32x16 bf16): lane<16 holds col N=lane, K 0-15 contiguous;
// lane>=16 holds K 16-31.  Caller passes p = col_base + (lane>>4)*16.
__device__ __forceinline__ v16bf frag_b_bf16(const __bf16* __restrict__ p) {
  v8bf lo = *(const v8bf*)p;
  v8bf hi = *(const v8bf*)(p + 8);
  v16bf f;
#pragma unroll
  for (int i = 0; i < 8; ++i) { f[i] = lo[i]; f[i + 8] = hi[i]; }
  return f;
}

__device__ __forceinline__ unsigned pack_bf16x2(float a, float b) {
  union { __bf16 h[2]; unsigned u; } p;
  p.h[0] = (__bf16)a; p.h[1] = (__bf16)b;
  return p.u;
}

// Async 16B global->LDS copy (CDNA5 ISA).  The low 32 bits
// of a generic pointer to a __shared__ object are the LDS byte offset.
__device__ __forceinline__ void async_copy16(const __bf16* g, __bf16* l) {
  const unsigned lofs = (unsigned)(unsigned long long)l;
  asm volatile("global_load_async_to_lds_b128 %0, %1, off"
               :: "v"(lofs), "v"((unsigned long long)g)
               : "memory");
}
__device__ __forceinline__ void wait_async0() {
  asm volatile("s_wait_asynccnt 0" ::: "memory");
}

// ---------------------------------------------------------------------------
// Kernel 0: f32 -> bf16 bulk conversion (vectorized 4-wide).
// ---------------------------------------------------------------------------
__global__ __launch_bounds__(256) void cvt_bf16_kernel(
    const float* __restrict__ s, __bf16* __restrict__ d, int n4) {
  const int i = blockIdx.x * 256 + threadIdx.x;
  if (i < n4) {
    const v4f f = ((const v4f*)s)[i];
    v4bf o;
#pragma unroll
    for (int j = 0; j < 4; ++j) o[j] = (__bf16)f[j];
    ((v4bf*)d)[i] = o;
  }
}

// ---------------------------------------------------------------------------
// Kernel 1: qkv = x @ w_qkv^T  (8192x1024 @ 1024x3072), wave tile 32x64.
// Waves of a block share the same 64-wide weight strip (B-fragment reuse).
// Scatter bf16: Q,K -> [bh][n][64]; V -> transposed [bh][64][n].
// ---------------------------------------------------------------------------
__global__ __launch_bounds__(256) void qkv_proj_kernel(
    const __bf16* __restrict__ xb, const __bf16* __restrict__ wq,
    __bf16* __restrict__ qbuf, __bf16* __restrict__ kbuf, __bf16* __restrict__ vT) {
  const int lane = threadIdx.x & 31;
  const int wid  = (blockIdx.x * blockDim.x + threadIdx.x) >> 5;
  const int tm   = wid % 256;           // 8192 / 32
  const int tn   = wid / 256;           // 3072 / 64
  const int m0   = tm * 32;
  const int n0   = tn * 64;
  const int lrow = lane & 15;
  const int hiA  = (lane >> 4) << 4;
  const int hiB  = (lane >> 4) << 4;

  v8f acc[2][4];
#pragma unroll
  for (int i = 0; i < 2; ++i)
#pragma unroll
    for (int j = 0; j < 4; ++j) acc[i][j] = (v8f)(0.f);

  for (int k0 = 0; k0 < 1024; k0 += 32) {
    v16bf a0 = frag_a_bf16(xb + (size_t)(m0 + lrow) * 1024 + k0 + hiA);
    v16bf a1 = frag_a_bf16(xb + (size_t)(m0 + 16 + lrow) * 1024 + k0 + hiA);
#pragma unroll
    for (int nt = 0; nt < 4; ++nt) {
      v16bf b = frag_b_bf16(wq + (size_t)(n0 + nt * 16 + lrow) * 1024 + k0 + hiB);
      acc[0][nt] = WMMA_BF16(a0, b, acc[0][nt]);
      acc[1][nt] = WMMA_BF16(a1, b, acc[1][nt]);
    }
  }

#pragma unroll
  for (int nt = 0; nt < 4; ++nt) {
    const int e = n0 + nt * 16 + lrow;       // row of w_qkv
    const int h = e / 192;                   // head
    const int r = e % 192;                   // 0-63 q, 64-127 k, 128-191 v
#pragma unroll
    for (int mt = 0; mt < 2; ++mt) {
#pragma unroll
      for (int rr = 0; rr < 8; ++rr) {
        const int t   = m0 + mt * 16 + ((lane >> 4) ? rr + 8 : rr);   // token
        const int bat = t >> 11;
        const int n   = t & 2047;
        const __bf16 v = (__bf16)acc[mt][nt][rr];
        const size_t bh = (size_t)(bat * 16 + h);
        if (r < 64)       qbuf[((bh * 2048 + n) << 6) + r] = v;
        else if (r < 128) kbuf[((bh * 2048 + n) << 6) + (r - 64)] = v;
        else              vT[((bh << 6) + (r - 128)) * 2048 + n] = v;
      }
    }
  }
}

// ---------------------------------------------------------------------------
// Kernel 2: flash attention, transposed, with double-buffered async LDS
// staging of the shared 64x64 K and V^T blocks.
//   S^T(keys x queries) = K . Q^T ; online softmax over keys (in-lane +
//   one xor-16 shuffle) ; out^T(d x queries) = V^T . P^T.
// ---------------------------------------------------------------------------
__device__ __forceinline__ void stage_kv(const __bf16* __restrict__ K,
                                         const __bf16* __restrict__ V, int kb,
                                         __bf16* kD, __bf16* vD, int tid) {
  // K block: rows kb..kb+63 of [n][64] are one contiguous 8 KB chunk.
  const int c0 = tid * 8;                 // element offset of 16B chunk
  const int c1 = (tid + 256) * 8;
  async_copy16(K + (size_t)kb * 64 + c0, kD + c0);
  async_copy16(K + (size_t)kb * 64 + c1, kD + c1);
  // V^T block: 64 rows (d) x 64 keys, row stride 2048.
  const int r0 = tid >> 3;                // 0..31
  const int cc = (tid & 7) * 8;
  const int r1 = r0 + 32;
  async_copy16(V + (size_t)r0 * 2048 + kb + cc, vD + r0 * 64 + cc);
  async_copy16(V + (size_t)r1 * 2048 + kb + cc, vD + r1 * 64 + cc);
}

__global__ __launch_bounds__(256) void attn_kernel(
    const __bf16* __restrict__ qbuf, const __bf16* __restrict__ kbuf,
    const __bf16* __restrict__ vT, __bf16* __restrict__ ctx) {
  __shared__ __bf16 kLds[2][64 * 64];     // 2 x 8 KB
  __shared__ __bf16 vLds[2][64 * 64];     // 2 x 8 KB

  const int tid  = threadIdx.x;
  const int lane = tid & 31;
  const int wv   = tid >> 5;
  const int bh   = blockIdx.x;                 // batch*16 + head
  const int bat  = bh >> 4, h = bh & 15;
  const int q0   = blockIdx.y * 128 + wv * 16; // query tile base
  const int lrow = lane & 15;
  const int hiA  = (lane >> 4) << 4;
  const int hiB  = (lane >> 4) << 4;

  const __bf16* Q = qbuf + (size_t)bh * 2048 * 64;
  const __bf16* K = kbuf + (size_t)bh * 2048 * 64;
  const __bf16* V = vT   + (size_t)bh * 64 * 2048;

  // Q^T B-operand fragments (persist across all key blocks): col = query lrow
  v16bf qf[2];
#pragma unroll
  for (int kd = 0; kd < 2; ++kd)
    qf[kd] = frag_b_bf16(Q + (size_t)(q0 + lrow) * 64 + kd * 32 + hiB);

  v8f vo[4];
#pragma unroll
  for (int i = 0; i < 4; ++i) vo[i] = (v8f)(0.f);
  float m_run = -__builtin_inff();
  float l_run = 0.f;

  stage_kv(K, V, 0, kLds[0], vLds[0], tid);    // prologue: stage first block

  for (int kb = 0; kb < 2048; kb += 64) {
    const int cur = (kb >> 6) & 1;
    wait_async0();                              // own copies for `cur` landed
    __syncthreads();                            // everyone's copies visible;
                                                // prev compute on cur^1 done
    if (kb + 64 < 2048)
      stage_kv(K, V, kb + 64, kLds[cur ^ 1], vLds[cur ^ 1], tid);
    const __bf16* kT = kLds[cur];
    const __bf16* vTile = vLds[cur];

    // ---- S^T = K . Q^T for 64 keys (4 key tiles x 2 K-steps) ----
    v8f st[4];
#pragma unroll
    for (int i = 0; i < 4; ++i) st[i] = (v8f)(0.f);
#pragma unroll
    for (int kt = 0; kt < 4; ++kt) {
      const __bf16* kp = kT + (kt * 16 + lrow) * 64 + hiA;
      st[kt] = WMMA_BF16(frag_a_bf16(kp),      qf[0], st[kt]);
      st[kt] = WMMA_BF16(frag_a_bf16(kp + 32), qf[1], st[kt]);
    }

    // ---- online softmax (scale = 1/sqrt(64)) ----
    float pm = -__builtin_inff();
#pragma unroll
    for (int kt = 0; kt < 4; ++kt)
#pragma unroll
      for (int r = 0; r < 8; ++r) {
        st[kt][r] *= 0.125f;
        pm = fmaxf(pm, st[kt][r]);
      }
    pm = fmaxf(pm, __shfl_xor(pm, 16, 32));     // combine lane halves
    const float m_new = fmaxf(m_run, pm);
    const float corr  = __expf(m_run - m_new);
    float pex[4][8];
    float psum = 0.f;
#pragma unroll
    for (int kt = 0; kt < 4; ++kt)
#pragma unroll
      for (int r = 0; r < 8; ++r) {
        const float p = __expf(st[kt][r] - m_new);
        pex[kt][r] = p;
        psum += p;
      }
    psum += __shfl_xor(psum, 16, 32);
    l_run = l_run * corr + psum;
    m_run = m_new;
#pragma unroll
    for (int dt = 0; dt < 4; ++dt)
#pragma unroll
      for (int r = 0; r < 8; ++r) vo[dt][r] *= corr;

    // ---- out^T += V^T . P^T  (S^T accumulator repacked as B operand) ----
#pragma unroll
    for (int kd2 = 0; kd2 < 2; ++kd2) {
      const int t0 = kd2 * 2, t1 = t0 + 1;
      unsigned own0[4], own1[4], recv[4];
#pragma unroll
      for (int j = 0; j < 4; ++j) {
        own0[j] = pack_bf16x2(pex[t0][2 * j], pex[t0][2 * j + 1]);
        own1[j] = pack_bf16x2(pex[t1][2 * j], pex[t1][2 * j + 1]);
        const unsigned send = (lane < 16) ? own1[j] : own0[j];
        recv[j] = (unsigned)__shfl_xor((int)send, 16, 32);
      }
      union { v16bf bf; unsigned u[8]; } pb;
#pragma unroll
      for (int j = 0; j < 4; ++j) {
        pb.u[j]     = (lane < 16) ? own0[j] : recv[j];
        pb.u[4 + j] = (lane < 16) ? recv[j] : own1[j];
      }
#pragma unroll
      for (int dt = 0; dt < 4; ++dt) {
        const __bf16* vp = vTile + (dt * 16 + lrow) * 64 + kd2 * 32 + hiA;
        vo[dt] = WMMA_BF16(frag_a_bf16(vp), pb.bf, vo[dt]);
      }
    }
  }

  // ---- normalize and store context (natural [b][n][h*64] layout, bf16) ----
  const float inv = 1.0f / l_run;
  const int  q    = q0 + lrow;
#pragma unroll
  for (int dt = 0; dt < 4; ++dt)
#pragma unroll
    for (int r = 0; r < 8; ++r) {
      const int d = dt * 16 + ((lane >> 4) ? r + 8 : r);
      ctx[(size_t)(bat * 2048 + q) * 1024 + h * 64 + d] = (__bf16)(vo[dt][r] * inv);
    }
}

// ---------------------------------------------------------------------------
// Kernel 3: y = ctx @ w_o^T  (8192x1024 @ 1024x1024), f32 output.
// ---------------------------------------------------------------------------
__global__ __launch_bounds__(256) void out_proj_kernel(
    const __bf16* __restrict__ ctx, const __bf16* __restrict__ wo,
    float* __restrict__ y) {
  const int lane = threadIdx.x & 31;
  const int wid  = (blockIdx.x * blockDim.x + threadIdx.x) >> 5;
  const int tm   = wid % 256;           // 8192 / 32
  const int tn   = wid / 256;           // 1024 / 64
  const int m0   = tm * 32;
  const int n0   = tn * 64;
  const int lrow = lane & 15;
  const int hiA  = (lane >> 4) << 4;
  const int hiB  = (lane >> 4) << 4;

  v8f acc[2][4];
#pragma unroll
  for (int i = 0; i < 2; ++i)
#pragma unroll
    for (int j = 0; j < 4; ++j) acc[i][j] = (v8f)(0.f);

  for (int k0 = 0; k0 < 1024; k0 += 32) {
    v16bf a0 = frag_a_bf16(ctx + (size_t)(m0 + lrow) * 1024 + k0 + hiA);
    v16bf a1 = frag_a_bf16(ctx + (size_t)(m0 + 16 + lrow) * 1024 + k0 + hiA);
#pragma unroll
    for (int nt = 0; nt < 4; ++nt) {
      v16bf b = frag_b_bf16(wo + (size_t)(n0 + nt * 16 + lrow) * 1024 + k0 + hiB);
      acc[0][nt] = WMMA_BF16(a0, b, acc[0][nt]);
      acc[1][nt] = WMMA_BF16(a1, b, acc[1][nt]);
    }
  }

#pragma unroll
  for (int nt = 0; nt < 4; ++nt) {
    const int e = n0 + nt * 16 + lrow;
#pragma unroll
    for (int mt = 0; mt < 2; ++mt)
#pragma unroll
      for (int rr = 0; rr < 8; ++rr) {
        const int t = m0 + mt * 16 + ((lane >> 4) ? rr + 8 : rr);
        y[(size_t)t * 1024 + e] = acc[mt][nt][rr];
      }
  }
}

// ---------------------------------------------------------------------------
extern "C" void kernel_launch(void* const* d_in, const int* in_sizes, int n_in,
                              void* d_out, int out_size, void* d_ws, size_t ws_size,
                              hipStream_t stream) {
  const float* x     = (const float*)d_in[0];   // (4,2048,1024)
  const float* w_qkv = (const float*)d_in[1];   // (3072,1024)
  const float* w_o   = (const float*)d_in[2];   // (1024,1024)
  float* out = (float*)d_out;

  char* ws = (char*)d_ws;
  const size_t MB = 1024 * 1024;
  __bf16* xb   = (__bf16*)(ws);             // 16 MB  [8192][1024]
  __bf16* wqb  = (__bf16*)(ws + 16 * MB);   //  6 MB  [3072][1024]
  __bf16* wob  = (__bf16*)(ws + 22 * MB);   //  2 MB  [1024][1024]
  __bf16* qbuf = (__bf16*)(ws + 24 * MB);   // 16 MB  [64][2048][64]
  __bf16* kbuf = (__bf16*)(ws + 40 * MB);   // 16 MB  [64][2048][64]
  __bf16* vT   = (__bf16*)(ws + 56 * MB);   // 16 MB  [64][64][2048]
  __bf16* ctx  = (__bf16*)(ws + 72 * MB);   // 16 MB  [8192][1024]

  // 0) one-pass precision conversion
  cvt_bf16_kernel<<<8192, 256, 0, stream>>>(x,     xb,  8192 * 1024 / 4);
  cvt_bf16_kernel<<<3072, 256, 0, stream>>>(w_qkv, wqb, 3072 * 1024 / 4);
  cvt_bf16_kernel<<<1024, 256, 0, stream>>>(w_o,   wob, 1024 * 1024 / 4);
  // 1) QKV projection: 256x48 wave tiles of 32x64, 8 waves/block
  qkv_proj_kernel<<<1536, 256, 0, stream>>>(xb, wqb, qbuf, kbuf, vT);
  // 2) attention: 64 (b,h) x 16 query-blocks of 128 (8 waves x 16 queries)
  attn_kernel<<<dim3(64, 16), 256, 0, stream>>>(qbuf, kbuf, vT, ctx);
  // 3) output projection: 256x16 wave tiles of 32x64
  out_proj_kernel<<<512, 256, 0, stream>>>(ctx, wob, out);
}
